// RNNPointerNetworkModel_10075993276791
// MI455X (gfx1250) — matrix operand order repaired and checked by reading, not compiled, not run on hardware
//
#include <hip/hip_runtime.h>
#include <hip/hip_fp16.h>
#include <math.h>

typedef __attribute__((ext_vector_type(16))) _Float16 v16h;
typedef __attribute__((ext_vector_type(8)))  _Float16 v8h;
typedef __attribute__((ext_vector_type(8)))  float    v8f;
typedef __attribute__((ext_vector_type(4)))  int      v4i;

constexpr int kB = 16, kS = 500, kT = 500, kTd = 499;
constexpr int kHID = 512, kHH = 256, kVOCAB = 5000, kMAXLEN = 500;

__device__ __forceinline__ float sigf(float x) { return 1.f / (1.f + __expf(-x)); }

// ---- async global->LDS staging (CDNA5 ASYNCcnt path), with safe fallback ----
#if defined(__has_builtin)
#if __has_builtin(__builtin_amdgcn_global_load_async_to_lds_b128) && \
    __has_builtin(__builtin_amdgcn_s_wait_asynccnt)
#define USE_ASYNC_LDS 1
#endif
#endif
#ifndef USE_ASYNC_LDS
#define USE_ASYNC_LDS 0
#endif

// copy 8 halves (16B) global -> LDS
__device__ __forceinline__ void lds_cp16(_Float16* dst, const _Float16* src) {
#if USE_ASYNC_LDS
    __builtin_amdgcn_global_load_async_to_lds_b128(
        (__attribute__((address_space(1))) v4i*)(src),
        (__attribute__((address_space(3))) v4i*)(dst), 0, 0);
#else
    *(v8h*)dst = *(const v8h*)src;
#endif
}
__device__ __forceinline__ void lds_cp_wait() {
#if USE_ASYNC_LDS
    __builtin_amdgcn_s_wait_asynccnt(0);
#endif
}

// ---- WMMA fragment loaders (16x16x32 f16, wave32) ---------------------------
// A (16x32, MxK): lane lo = row M; two contiguous 8-half chunks at
//   k = hi*8..+7 and k = 16+hi*8..+7  (ISA 7.12.2, 16-bit A layout)
__device__ __forceinline__ v16h load_a_frag(const _Float16* base, int ldk, int lane) {
    int lo = lane & 15, hi = lane >> 4;
    const _Float16* p = base + (long long)lo * ldk + hi * 8;
    v8h x0 = *(const v8h*)(p);
    v8h x1 = *(const v8h*)(p + 16);
    return __builtin_shufflevector(x0, x1, 0,1,2,3,4,5,6,7,8,9,10,11,12,13,14,15);
}
// B (32x16, KxN) staged in LDS row-major [n][ldk]: lane lo = col N,
//   two 8-half chunks at k = hi*8..+7 and k = 16+hi*8..+7 (matches A)
__device__ __forceinline__ v16h load_b_frag(const _Float16* base, int ldk, int lane) {
    int lo = lane & 15, hi = lane >> 4;
    const _Float16* p = base + (long long)lo * ldk + hi * 8;
    v8h x0 = *(const v8h*)(p);
    v8h x1 = *(const v8h*)(p + 16);
    return __builtin_shufflevector(x0, x1, 0,1,2,3,4,5,6,7,8,9,10,11,12,13,14,15);
}

// ---- Generic batched WMMA GEMM: C[M,N] = act(A[M,K] @ op(B) + bias) ---------
// BL==0: B stored [N,K] row-major (TN: W^T weights / row-major key matrix)
// BL==1: B stored [K,N] row-major (NN: attention context)
// ACT: 0 none, 1 relu, 2 tanh. C (f32) and/or Ch (f16) optional.
template <int ACT, int BL>
__global__ __launch_bounds__(256) void gemm_wmma(
    const _Float16* __restrict__ A, long long sA,
    const _Float16* __restrict__ Bm, long long sB,
    const float* __restrict__ bias,
    float* __restrict__ C, long long sC,
    _Float16* __restrict__ Ch, long long sCh,
    int M, int N, int K) {
    __shared__ _Float16 As[32 * 64];   // 32 rows x 64 k
    __shared__ _Float16 Bs[64 * 64];   // 64 cols x 64 k
    const int bz = blockIdx.z;
    A += (long long)bz * sA;
    Bm += (long long)bz * sB;
    const int tid = threadIdx.x;
    const int lane = tid & 31, wave = tid >> 5;
    const int mi = wave >> 2, ni = wave & 3;      // 2x4 wave grid -> 32x64 tile
    const int m0 = blockIdx.y * 32, n0 = blockIdx.x * 64;
    const int mrow = m0 + mi * 16;
    v8f acc = {};

    for (int k0 = 0; k0 < K; k0 += 64) {
        // ---- stage A tile 32x64: 256 threads x 8 halves ----
        {
            int r = tid >> 3, c8 = (tid & 7) * 8;
            int gr = m0 + r;
            if ((m0 + 32 <= M) && (k0 + 64 <= K)) {        // uniform fast path
                lds_cp16(As + r * 64 + c8, A + (long long)gr * K + k0 + c8);
            } else {
                #pragma unroll
                for (int j = 0; j < 8; ++j) {
                    int kk = k0 + c8 + j;
                    As[r * 64 + c8 + j] =
                        (gr < M && kk < K) ? A[(long long)gr * K + kk] : (_Float16)0.f;
                }
            }
        }
        // ---- stage B tile 64x64 ----
        if (BL == 0) {
            int r = tid >> 2, c16 = (tid & 3) * 16;
            int gn = n0 + r;
            if ((n0 + 64 <= N) && (k0 + 64 <= K)) {        // uniform fast path
                const _Float16* src = Bm + (long long)gn * K + k0 + c16;
                lds_cp16(Bs + r * 64 + c16, src);
                lds_cp16(Bs + r * 64 + c16 + 8, src + 8);
            } else {
                #pragma unroll
                for (int j = 0; j < 16; ++j) {
                    int kk = k0 + c16 + j;
                    Bs[r * 64 + c16 + j] =
                        (gn < N && kk < K) ? Bm[(long long)gn * K + kk] : (_Float16)0.f;
                }
            }
        } else { // transpose-gather from [K,N]
            #pragma unroll
            for (int j = 0; j < 16; ++j) {
                int e = tid * 16 + j;
                int nn = e >> 6, kk = e & 63;
                int gk = k0 + kk, gn = n0 + nn;
                Bs[nn * 64 + kk] =
                    (gk < K && gn < N) ? Bm[(long long)gk * N + gn] : (_Float16)0.f;
            }
        }
        lds_cp_wait();
        __syncthreads();
        #pragma unroll
        for (int kb = 0; kb < 64; kb += 32) {
            v16h af = load_a_frag(As + mi * 16 * 64 + kb, 64, lane);
            v16h bf = load_b_frag(Bs + ni * 16 * 64 + kb, 64, lane);
            acc = __builtin_amdgcn_wmma_f32_16x16x32_f16(false, af, false, bf,
                                                         (short)0, acc, false, false);
        }
        __syncthreads();
    }
    // epilogue: VGPR r -> row = mrow + r + 8*hi, col = n0 + ni*16 + (lane&15)
    int lo = lane & 15, hi = lane >> 4;
    int col = n0 + ni * 16 + lo;
    if ((mrow + 16 <= M) && (n0 + 64 <= N)) {              // uniform fast path
        float bv = bias ? bias[col] : 0.f;
        #pragma unroll
        for (int r = 0; r < 8; ++r) {
            int row = mrow + r + 8 * hi;
            float v = acc[r] + bv;
            if (ACT == 1) v = fmaxf(v, 0.f);
            else if (ACT == 2) v = tanhf(v);
            long long o = (long long)row * N + col;
            if (C)  C[(long long)bz * sC + o] = v;
            if (Ch) Ch[(long long)bz * sCh + o] = (_Float16)v;
        }
    } else if (col < N) {
        float bv = bias ? bias[col] : 0.f;
        #pragma unroll
        for (int r = 0; r < 8; ++r) {
            int row = mrow + r + 8 * hi;
            if (row < M) {
                float v = acc[r] + bv;
                if (ACT == 1) v = fmaxf(v, 0.f);
                else if (ACT == 2) v = tanhf(v);
                long long o = (long long)row * N + col;
                if (C)  C[(long long)bz * sC + o] = v;
                if (Ch) Ch[(long long)bz * sCh + o] = (_Float16)v;
            }
        }
    }
}

// ---- Persistent LSTM recurrence: one workgroup, 32 waves, state in LDS ------
// xg[b,t,4H] precomputed (x@Wih^T + bih + bhh). Per step: g = xg[t] + h@Whh^T,
// gate update, h/c kept in LDS. Whh (f16, [4H,H]) staged 32-K-slabs into LDS.
template <int H>
__global__ __launch_bounds__(1024) void lstm_rec(
    const float* __restrict__ xg, const _Float16* __restrict__ Whh,
    const float* __restrict__ h0, const float* __restrict__ c0,
    float* __restrict__ y, _Float16* __restrict__ yh, int ldy, int coly,
    float* __restrict__ hT, float* __restrict__ cT, int ldh, int colh,
    int Tn, int reverse) {
    extern __shared__ char smem[];
    _Float16* hf  = (_Float16*)smem;                                   // 16*H f16
    float*    cst = (float*)(smem + (size_t)16 * H * sizeof(_Float16));// 16*H f32
    float*    g   = cst + 16 * H;                                      // 16*4H f32
    _Float16* Bs  = (_Float16*)(g + 16 * 4 * H);                       // 4H*32 f16

    constexpr int NT  = (4 * H) / 16;       // N tiles (64 or 128)
    constexpr int TPW = NT / 32;            // tiles per wave (2 or 4)
    constexpr int CPT = (16 * H) / 1024;    // state cells per thread (4 or 8)
    constexpr int NCH = (4 * H * 32) / (8 * 1024); // 8-half chunks per thread

    const int tid = threadIdx.x;
    const int lane = tid & 31, wave = tid >> 5;

    for (int j = 0; j < CPT; ++j) {
        int cell = tid + j * 1024;
        hf[cell]  = (_Float16)(h0 ? h0[cell] : 0.f);
        cst[cell] = c0 ? c0[cell] : 0.f;
    }
    __syncthreads();

    for (int tt = 0; tt < Tn; ++tt) {
        const int t = reverse ? (Tn - 1 - tt) : tt;
        v8f acc[TPW];
        #pragma unroll
        for (int p = 0; p < TPW; ++p) acc[p] = {};

        for (int ks = 0; ks < H / 32; ++ks) {
            #pragma unroll
            for (int j = 0; j < NCH; ++j) { // stage Whh[:, ks*32 .. +31]
                int e = (tid + j * 1024) * 8;
                int n = e >> 5, kk = e & 31;
                lds_cp16(Bs + n * 32 + kk, Whh + (long long)n * H + ks * 32 + kk);
            }
            lds_cp_wait();
            __syncthreads();
            v16h af = load_a_frag(hf + ks * 32, H, lane);
            #pragma unroll
            for (int p = 0; p < TPW; ++p) {
                v16h bf = load_b_frag(Bs + (wave * TPW + p) * 16 * 32, 32, lane);
                acc[p] = __builtin_amdgcn_wmma_f32_16x16x32_f16(
                    false, af, false, bf, (short)0, acc[p], false, false);
            }
            __syncthreads();
        }
        { // scatter accumulators into the gate buffer
            int lo = lane & 15, hi = lane >> 4;
            #pragma unroll
            for (int p = 0; p < TPW; ++p) {
                int col = (wave * TPW + p) * 16 + lo;
                #pragma unroll
                for (int r = 0; r < 8; ++r) g[(r + 8 * hi) * 4 * H + col] = acc[p][r];
            }
        }
        __syncthreads();
        for (int j = 0; j < CPT; ++j) { // gate nonlinearity + state update
            int cell = tid + j * 1024;
            int b = cell / H, u = cell % H;
            const float* xr = xg + ((long long)b * Tn + t) * 4 * H;
            float gi = g[b * 4 * H + u]           + xr[u];
            float gf = g[b * 4 * H + H + u]       + xr[H + u];
            float gg = g[b * 4 * H + 2 * H + u]   + xr[2 * H + u];
            float go = g[b * 4 * H + 3 * H + u]   + xr[3 * H + u];
            float cn = sigf(gf) * cst[cell] + sigf(gi) * tanhf(gg);
            float hn = sigf(go) * tanhf(cn);
            cst[cell] = cn;
            hf[cell]  = (_Float16)hn;
            long long yi = ((long long)b * Tn + t) * ldy + coly + u;
            y[yi]  = hn;
            yh[yi] = (_Float16)hn;
        }
        __syncthreads();
    }
    if (hT) {
        for (int j = 0; j < CPT; ++j) {
            int cell = tid + j * 1024;
            int b = cell / H, u = cell % H;
            hT[b * ldh + colh + u] = (float)hf[cell];
            cT[b * ldh + colh + u] = cst[cell];
        }
    }
}

// ---- Small elementwise / reduction kernels ----------------------------------
__global__ void f2h_kernel(const float* __restrict__ s, _Float16* __restrict__ d, long long n) {
    long long i = (long long)blockIdx.x * blockDim.x + threadIdx.x;
    long long st = (long long)gridDim.x * blockDim.x;
    for (; i < n; i += st) d[i] = (_Float16)s[i];
}
__global__ void addv_kernel(const float* a, const float* b, float* c, int n) {
    int i = blockIdx.x * blockDim.x + threadIdx.x;
    if (i < n) c[i] = a[i] + b[i];
}
__global__ void embed_enc_kernel(const int* __restrict__ inputs, const unsigned char* __restrict__ mask,
                                 const float* __restrict__ tok_emb, const float* __restrict__ pos_emb,
                                 float* __restrict__ pos_only, _Float16* __restrict__ pos_only_h,
                                 float* __restrict__ x, _Float16* __restrict__ xh) {
    int bs = blockIdx.x, t = threadIdx.x;
    int s = bs % kS;
    int pidx = mask[bs] ? s : kMAXLEN;
    int tok = inputs[bs];
    float pv = pos_emb[(long long)pidx * kHH + t];
    float tv = tok_emb[(long long)tok * kHH + t];
    pos_only[(long long)bs * kHH + t] = pv;
    pos_only_h[(long long)bs * kHH + t] = (_Float16)pv;
    long long xb = (long long)bs * kHID;
    x[xb + t] = pv;          x[xb + kHH + t] = tv;
    xh[xb + t] = (_Float16)pv; xh[xb + kHH + t] = (_Float16)tv;
}
__global__ void embed_dec_kernel(const int* __restrict__ output, const unsigned char* __restrict__ omask,
                                 const float* __restrict__ tok_emb, const float* __restrict__ pos_emb,
                                 float* __restrict__ x, _Float16* __restrict__ xh) {
    int bt = blockIdx.x, th = threadIdx.x;
    int b = bt / kTd, t = bt % kTd;
    int src = b * kT + t;
    int pidx = omask[src] ? t : kMAXLEN;
    int tok = output[src];
    float pv = pos_emb[(long long)pidx * kHH + th];
    float tv = tok_emb[(long long)tok * kHH + th];
    long long xb = (long long)bt * kHID;
    x[xb + th] = pv;           x[xb + kHH + th] = tv;
    xh[xb + th] = (_Float16)pv; xh[xb + kHH + th] = (_Float16)tv;
}
__global__ void softmax_rows_kernel(const float* __restrict__ sc, const unsigned char* __restrict__ imask,
                                    _Float16* __restrict__ awh) {
    __shared__ float red[512];
    int bt = blockIdx.x, tid = threadIdx.x;
    int b = bt / kTd;
    const float* row = sc + (long long)bt * kS;
    const unsigned char* m = imask + b * kS;
    float mx = -INFINITY;
    for (int s = tid; s < kS; s += 512) if (m[s]) mx = fmaxf(mx, row[s]);
    red[tid] = mx; __syncthreads();
    for (int o = 256; o > 0; o >>= 1) { if (tid < o) red[tid] = fmaxf(red[tid], red[tid + o]); __syncthreads(); }
    mx = red[0]; __syncthreads();
    float sum = 0.f;
    for (int s = tid; s < kS; s += 512) if (m[s]) sum += __expf(row[s] - mx);
    red[tid] = sum; __syncthreads();
    for (int o = 256; o > 0; o >>= 1) { if (tid < o) red[tid] += red[tid + o]; __syncthreads(); }
    float inv = 1.f / red[0];
    for (int s = tid; s < kS; s += 512) {
        float v = m[s] ? __expf(row[s] - mx) * inv : 0.f;
        awh[(long long)bt * kS + s] = (_Float16)v;
    }
}
__global__ void cat_kernel(const float* __restrict__ ctx, const float* __restrict__ dx,
                           _Float16* __restrict__ cat) {
    long long n = (long long)kB * kTd * 1024;
    long long i = (long long)blockIdx.x * blockDim.x + threadIdx.x;
    long long st = (long long)gridDim.x * blockDim.x;
    for (; i < n; i += st) {
        long long bt = i >> 10; int j = (int)(i & 1023);
        float v = (j < kHID) ? ctx[bt * kHID + j] : dx[bt * kHID + (j - kHID)];
        cat[i] = (_Float16)v;
    }
}
__global__ void maskdm_kernel(float* __restrict__ dec_out, _Float16* __restrict__ dec_out_h,
                              const unsigned char* __restrict__ omask) {
    long long n = (long long)kB * kTd * kHID;
    long long i = (long long)blockIdx.x * blockDim.x + threadIdx.x;
    long long st = (long long)gridDim.x * blockDim.x;
    for (; i < n; i += st) {
        long long bt = i / kHID;
        int b = (int)(bt / kTd), t = (int)(bt % kTd);
        float v = dec_out[i];
        if (!omask[b * kT + t + 1]) v = 0.f;
        dec_out[i] = v;
        dec_out_h[i] = (_Float16)v;
    }
}
__global__ void copyhead_kernel(const _Float16* __restrict__ t1, const float* __restrict__ W2,
                                const float* __restrict__ b2, float* __restrict__ out) {
    __shared__ float red[256];
    int row = blockIdx.x, tid = threadIdx.x;
    float s = 0.f;
    for (int k = tid; k < kHID; k += 256) s += (float)t1[(long long)row * kHID + k] * W2[k];
    red[tid] = s; __syncthreads();
    for (int o = 128; o > 0; o >>= 1) { if (tid < o) red[tid] += red[tid + o]; __syncthreads(); }
    if (tid == 0) out[row] = 1.f / (1.f + __expf(-(red[0] + b2[0])));
}
__global__ void ptrmask_kernel(float* __restrict__ po, const unsigned char* __restrict__ imask) {
    long long n = (long long)kB * kTd * kS;
    long long i = (long long)blockIdx.x * blockDim.x + threadIdx.x;
    long long st = (long long)gridDim.x * blockDim.x;
    for (; i < n; i += st) {
        int s = (int)(i % kS);
        long long bt = i / kS;
        int b = (int)(bt / kTd);
        if (!imask[b * kS + s]) po[i] = -INFINITY;
    }
}

// ---- Host orchestration -----------------------------------------------------
static void launch_gemm(int act, int bl,
                        const _Float16* A, long long sA, const _Float16* Bm, long long sB,
                        const float* bias, float* C, long long sC, _Float16* Ch, long long sCh,
                        int M, int N, int K, int batch, hipStream_t stream) {
    dim3 grid((N + 63) / 64, (M + 31) / 32, batch);
    if (bl == 1)          gemm_wmma<0, 1><<<grid, 256, 0, stream>>>(A, sA, Bm, sB, bias, C, sC, Ch, sCh, M, N, K);
    else if (act == 1)    gemm_wmma<1, 0><<<grid, 256, 0, stream>>>(A, sA, Bm, sB, bias, C, sC, Ch, sCh, M, N, K);
    else if (act == 2)    gemm_wmma<2, 0><<<grid, 256, 0, stream>>>(A, sA, Bm, sB, bias, C, sC, Ch, sCh, M, N, K);
    else                  gemm_wmma<0, 0><<<grid, 256, 0, stream>>>(A, sA, Bm, sB, bias, C, sC, Ch, sCh, M, N, K);
}

extern "C" void kernel_launch(void* const* d_in, const int* in_sizes, int n_in,
                              void* d_out, int out_size, void* d_ws, size_t ws_size,
                              hipStream_t stream) {
    (void)in_sizes; (void)n_in; (void)out_size; (void)ws_size;
    const int*           inputs  = (const int*)d_in[0];
    const unsigned char* imask   = (const unsigned char*)d_in[1];
    const int*           output  = (const int*)d_in[2];
    const unsigned char* omask   = (const unsigned char*)d_in[3];
    const float* enc_tok = (const float*)d_in[4];
    const float* enc_pos = (const float*)d_in[5];
    const float* dec_tok = (const float*)d_in[6];
    const float* dec_pos = (const float*)d_in[7];
    const float* attn_W  = (const float*)d_in[8];
    const float* attn_b  = (const float*)d_in[9];
    const float* copy_W1 = (const float*)d_in[10];
    const float* copy_b1 = (const float*)d_in[11];
    const float* copy_W2 = (const float*)d_in[12];
    const float* copy_b2 = (const float*)d_in[13];
    const float* ptr_W1  = (const float*)d_in[14];
    const float* ptr_b1  = (const float*)d_in[15];
    const float* ptr_W2  = (const float*)d_in[16];
    const float* ptr_b2  = (const float*)d_in[17];
    const float* out_W1  = (const float*)d_in[18];
    const float* out_b1  = (const float*)d_in[19];
    const float* out_W2  = (const float*)d_in[20];
    const float* out_b2  = (const float*)d_in[21];
    const float *encWih[4], *encWhh[4], *encBih[4], *encBhh[4];
    const float *decWih[2], *decWhh[2], *decBih[2], *decBhh[2];
    int idx = 22;
    for (int l = 0; l < 2; ++l) {
        for (int d = 0; d < 2; ++d) {
            encWih[l * 2 + d] = (const float*)d_in[idx++];
            encWhh[l * 2 + d] = (const float*)d_in[idx++];
            encBih[l * 2 + d] = (const float*)d_in[idx++];
            encBhh[l * 2 + d] = (const float*)d_in[idx++];
        }
        decWih[l] = (const float*)d_in[idx++];
        decWhh[l] = (const float*)d_in[idx++];
        decBih[l] = (const float*)d_in[idx++];
        decBhh[l] = (const float*)d_in[idx++];
    }

    // ---- bump allocator over d_ws ----
    char* wp = (char*)d_ws;
    auto alloc = [&](size_t bytes) -> void* {
        void* p = (void*)wp;
        wp += (bytes + 255) & ~(size_t)255;
        return p;
    };
    auto f2h = [&](const float* src, _Float16* dst, long long n) {
        int blocks = (int)((n + 255) / 256); if (blocks > 4096) blocks = 4096;
        f2h_kernel<<<blocks, 256, 0, stream>>>(src, dst, n);
    };

    _Float16 *encWih_h[4], *encWhh_h[4], *decWih_h[2], *decWhh_h[2];
    float *encBias[4], *decBias[2];
    for (int i = 0; i < 4; ++i) {
        encWih_h[i] = (_Float16*)alloc((size_t)4 * kHH * kHID * 2);
        encWhh_h[i] = (_Float16*)alloc((size_t)4 * kHH * kHH * 2);
        encBias[i]  = (float*)alloc((size_t)4 * kHH * 4);
    }
    for (int i = 0; i < 2; ++i) {
        decWih_h[i] = (_Float16*)alloc((size_t)4 * kHID * kHID * 2);
        decWhh_h[i] = (_Float16*)alloc((size_t)4 * kHID * kHID * 2);
        decBias[i]  = (float*)alloc((size_t)4 * kHID * 4);
    }
    _Float16* attnW_h  = (_Float16*)alloc((size_t)kHID * 2 * kHID * 2);
    _Float16* copyW1_h = (_Float16*)alloc((size_t)kHID * kHID * 2);
    _Float16* ptrW1_h  = (_Float16*)alloc((size_t)kHID * kHID * 2);
    _Float16* ptrW2_h  = (_Float16*)alloc((size_t)kHH * kHID * 2);
    _Float16* outW1_h  = (_Float16*)alloc((size_t)kHID * kHID * 2);
    _Float16* outW2_h  = (_Float16*)alloc((size_t)kVOCAB * kHID * 2);

    float*    pos_only   = (float*)alloc((size_t)kB * kS * kHH * 4);
    _Float16* pos_only_h = (_Float16*)alloc((size_t)kB * kS * kHH * 2);
    float*    encX   = (float*)alloc((size_t)kB * kS * kHID * 4);
    _Float16* encX_h = (_Float16*)alloc((size_t)kB * kS * kHID * 2);
    float*    encY   = (float*)alloc((size_t)kB * kS * kHID * 4);
    _Float16* encY_h = (_Float16*)alloc((size_t)kB * kS * kHID * 2);
    float*    xg     = (float*)alloc((size_t)kB * kS * 4 * kHID * 4); // also attn scratch later
    float*    dec_h0 = (float*)alloc((size_t)2 * kB * kHID * 4);
    float*    dec_c0 = (float*)alloc((size_t)2 * kB * kHID * 4);
    float*    decX   = (float*)alloc((size_t)kB * kTd * kHID * 4);
    _Float16* decX_h = (_Float16*)alloc((size_t)kB * kTd * kHID * 2);
    float*    decY   = (float*)alloc((size_t)kB * kTd * kHID * 4);
    _Float16* decY_h = (_Float16*)alloc((size_t)kB * kTd * kHID * 2);
    float*    t1     = (float*)alloc((size_t)kB * kTd * kHID * 4);
    _Float16* t1_h   = (_Float16*)alloc((size_t)kB * kTd * kHID * 2);
    float*    dec_out   = (float*)alloc((size_t)kB * kTd * kHID * 4);
    _Float16* dec_out_h = (_Float16*)alloc((size_t)kB * kTd * kHID * 2);
    _Float16* ptrff_h   = (_Float16*)alloc((size_t)kB * kTd * kHH * 2);

    // attention buffers alias the xg region (xg is dead after the decoder)
    char* ap = (char*)xg;
    float*    scores = (float*)ap;                 ap += ((size_t)kB * kTd * kS * 4 + 255) & ~(size_t)255;
    _Float16* aw_h   = (_Float16*)ap;              ap += ((size_t)kB * kTd * kS * 2 + 255) & ~(size_t)255;
    float*    ctx    = (float*)ap;                 ap += ((size_t)kB * kTd * kHID * 4 + 255) & ~(size_t)255;
    _Float16* cat_h  = (_Float16*)ap;

    // ---- weight conversion + fused biases ----
    for (int i = 0; i < 4; ++i) {
        f2h(encWih[i], encWih_h[i], (long long)4 * kHH * kHID);
        f2h(encWhh[i], encWhh_h[i], (long long)4 * kHH * kHH);
        addv_kernel<<<(4 * kHH + 255) / 256, 256, 0, stream>>>(encBih[i], encBhh[i], encBias[i], 4 * kHH);
    }
    for (int i = 0; i < 2; ++i) {
        f2h(decWih[i], decWih_h[i], (long long)4 * kHID * kHID);
        f2h(decWhh[i], decWhh_h[i], (long long)4 * kHID * kHID);
        addv_kernel<<<(4 * kHID + 255) / 256, 256, 0, stream>>>(decBih[i], decBhh[i], decBias[i], 4 * kHID);
    }
    f2h(attn_W, attnW_h, (long long)kHID * 2 * kHID);
    f2h(copy_W1, copyW1_h, (long long)kHID * kHID);
    f2h(ptr_W1, ptrW1_h, (long long)kHID * kHID);
    f2h(ptr_W2, ptrW2_h, (long long)kHH * kHID);
    f2h(out_W1, outW1_h, (long long)kHID * kHID);
    f2h(out_W2, outW2_h, (long long)kVOCAB * kHID);

    // ---- encoder ----
    embed_enc_kernel<<<kB * kS, kHH, 0, stream>>>(inputs, imask, enc_tok, enc_pos,
                                                  pos_only, pos_only_h, encX, encX_h);
    const size_t shEnc = (size_t)16 * kHH * 2 + (size_t)16 * kHH * 4 +
                         (size_t)16 * 4 * kHH * 4 + (size_t)4 * kHH * 32 * 2;   // 152 KB
    const size_t shDec = (size_t)16 * kHID * 2 + (size_t)16 * kHID * 4 +
                         (size_t)16 * 4 * kHID * 4 + (size_t)4 * kHID * 32 * 2; // 304 KB
    float*    exf[2] = {encX, encY};
    _Float16* exh[2] = {encX_h, encY_h};
    int cur = 0;
    for (int l = 0; l < 2; ++l) {
        int nxt = cur ^ 1;
        for (int d = 0; d < 2; ++d) {
            launch_gemm(0, 0, exh[cur], 0, encWih_h[l * 2 + d], 0, encBias[l * 2 + d],
                        xg, 0, nullptr, 0, kB * kS, 4 * kHH, kHID, 1, stream);
            lstm_rec<kHH><<<1, 1024, shEnc, stream>>>(
                xg, encWhh_h[l * 2 + d], nullptr, nullptr,
                exf[nxt], exh[nxt], kHID, d * kHH,
                dec_h0 + (size_t)l * kB * kHID, dec_c0 + (size_t)l * kB * kHID, kHID, d * kHH,
                kS, d);
        }
        cur = nxt;
    }
    float*    enc_val   = exf[cur];
    _Float16* enc_val_h = exh[cur];
    (void)enc_val;

    // ---- decoder ----
    embed_dec_kernel<<<kB * kTd, kHH, 0, stream>>>(output, omask, dec_tok, dec_pos, decX, decX_h);
    float*    dxf[2] = {decX, decY};
    _Float16* dxh[2] = {decX_h, decY_h};
    int dc = 0;
    for (int l = 0; l < 2; ++l) {
        launch_gemm(0, 0, dxh[dc], 0, decWih_h[l], 0, decBias[l],
                    xg, 0, nullptr, 0, kB * kTd, 4 * kHID, kHID, 1, stream);
        lstm_rec<kHID><<<1, 1024, shDec, stream>>>(
            xg, decWhh_h[l], dec_h0 + (size_t)l * kB * kHID, dec_c0 + (size_t)l * kB * kHID,
            dxf[dc ^ 1], dxh[dc ^ 1], kHID, 0, nullptr, nullptr, 0, 0, kTd, 0);
        dc ^= 1;
    }
    float*    dx   = dxf[dc];
    _Float16* dx_h = dxh[dc];

    // ---- attention ----
    launch_gemm(0, 0, dx_h, (long long)kTd * kHID, enc_val_h, (long long)kS * kHID, nullptr,
                scores, (long long)kTd * kS, nullptr, 0, kTd, kS, kHID, kB, stream);
    softmax_rows_kernel<<<kB * kTd, 512, 0, stream>>>(scores, imask, aw_h);
    launch_gemm(0, 1, aw_h, (long long)kTd * kS, enc_val_h, (long long)kS * kHID, nullptr,
                ctx, (long long)kTd * kHID, nullptr, 0, kTd, kHID, kS, kB, stream);
    cat_kernel<<<4096, 256, 0, stream>>>(ctx, dx, cat_h);
    launch_gemm(2, 0, cat_h, 0, attnW_h, 0, attn_b,
                dec_out, 0, nullptr, 0, kB * kTd, kHID, 2 * kHID, 1, stream);
    maskdm_kernel<<<4096, 256, 0, stream>>>(dec_out, dec_out_h, omask);

    float* out_iscopy = (float*)d_out;
    float* out_value  = out_iscopy + (long long)kB * kTd;
    float* out_ptr    = out_value + (long long)kB * kTd * kVOCAB;

    // ---- copy head (sigmoid scalar) ----
    launch_gemm(1, 0, dec_out_h, 0, copyW1_h, 0, copy_b1, t1, 0, t1_h, 0,
                kB * kTd, kHID, kHID, 1, stream);
    copyhead_kernel<<<kB * kTd, 256, 0, stream>>>(t1_h, copy_W2, copy_b2, out_iscopy);

    // ---- value head (vocab logits, straight into d_out) ----
    launch_gemm(1, 0, dec_out_h, 0, outW1_h, 0, out_b1, t1, 0, t1_h, 0,
                kB * kTd, kHID, kHID, 1, stream);
    launch_gemm(0, 0, t1_h, 0, outW2_h, 0, out_b2, out_value, 0, nullptr, 0,
                kB * kTd, kVOCAB, kHID, 1, stream);

    // ---- pointer head ----
    launch_gemm(1, 0, dec_out_h, 0, ptrW1_h, 0, ptr_b1, t1, 0, t1_h, 0,
                kB * kTd, kHID, kHID, 1, stream);
    launch_gemm(0, 0, t1_h, 0, ptrW2_h, 0, ptr_b2, nullptr, 0, ptrff_h, 0,
                kB * kTd, kHH, kHID, 1, stream);
    launch_gemm(0, 0, ptrff_h, (long long)kTd * kHH, pos_only_h, (long long)kS * kHH, nullptr,
                out_ptr, (long long)kTd * kS, nullptr, 0, kTd, kS, kHH, kB, stream);
    ptrmask_kernel<<<4096, 256, 0, stream>>>(out_ptr, imask);
}
